// Block_9637906612952
// MI455X (gfx1250) — compile-verified, matrix operand order checked by reading, not run on hardware
//
#include <hip/hip_runtime.h>
#include <hip/hip_bf16.h>
#include <math.h>
#include <stdint.h>

// ---------------------------------------------------------------------------
// Transformer block (pre-norm attn + MLP, LayerScale) for MI455X / gfx1250.
// GEMMs: v_wmma_f32_16x16x32_bf16 (fp32 accumulate), wave32.
// Staging: global_load_async_to_lds_b128 (ASYNCcnt path, no VGPR round-trip).
// B=16, N=1024, C=768, H=12, HD=64, MLP=3072.
// ---------------------------------------------------------------------------

#define Bq   16
#define Nn   1024
#define Cc   768
#define Hh   12
#define HD   64
#define MLPd 3072
#define Mrows (Bq * Nn)          // 16384

#define USE_ASYNC_STAGING 1

typedef __attribute__((ext_vector_type(16))) __bf16 v16bf;
typedef __attribute__((ext_vector_type(8)))  float  v8f;

// ---------------------------------------------------------------------------
// Async global -> LDS 128-bit copy (GLOBAL_LOAD_ASYNC_TO_LDS_B128, GV mode).
// vdst = per-lane LDS byte offset, vaddr = per-lane 64-bit global address.
// Generic LDS pointers keep the LDS byte offset in bits [31:0].
// ---------------------------------------------------------------------------
__device__ __forceinline__ void async_b128(void* lds, const void* g) {
#if USE_ASYNC_STAGING
  unsigned lds_off = (unsigned)(uintptr_t)lds;
  unsigned long long ga = (unsigned long long)(uintptr_t)g;
  asm volatile("global_load_async_to_lds_b128 %0, %1, off"
               :: "v"(lds_off), "v"(ga) : "memory");
#else
  *(uint4*)lds = *(const uint4*)g;
#endif
}

__device__ __forceinline__ void async_wait0() {
#if USE_ASYNC_STAGING
  asm volatile("s_wait_asynccnt 0" ::: "memory");
#endif
}

// Load one 16x32 bf16 fragment (A-layout; also used for B when the tile is
// staged [N][K] row-major in LDS). Per CDNA5 ISA 16-bit A-matrix layout:
//   lane L: row = ridx (caller passes m0 + (lane&15)), half = lane>>4
//   VGPR v<4:  K = 8*half + 2v, 2v+1        -> dword  4*half + v
//   VGPR v>=4: K = 16 + 8*half + 2(v-4), +1 -> dword  8 + 4*half + (v-4)
__device__ __forceinline__ v16bf frag_ld(const __bf16* lds, int ridx,
                                         int stride_e, int kbase_e, int half) {
  union { v16bf v; unsigned int u[8]; } f;
  const unsigned int* p = (const unsigned int*)(lds + ridx * stride_e + kbase_e);
  const int kb = 4 * half;
#pragma unroll
  for (int t = 0; t < 4; ++t) f.u[t] = p[kb + t];
#pragma unroll
  for (int t = 0; t < 4; ++t) f.u[4 + t] = p[8 + kb + t];
  return f.v;
}

__device__ __forceinline__ v8f wmma_bf16(v16bf a, v16bf b, v8f c) {
  return __builtin_amdgcn_wmma_f32_16x16x32_bf16(
      /*neg_a=*/false, a, /*neg_b=*/false, b,
      /*c_mod=*/(short)0, c, /*reuse_a=*/false, /*reuse_b=*/false);
}

// ---------------------------------------------------------------------------
// Weight convert + transpose: W[K][Nc] fp32  ->  Wt[Nc][K] bf16.
// 64x64 LDS tile; coalesced read along n, coalesced write along k.
// Makes every GEMM weight tile a contiguous [n][k] row -> async b128 copies.
// ---------------------------------------------------------------------------
__global__ __launch_bounds__(256)
void cvt_transpose_bf16(const float* __restrict__ w, __bf16* __restrict__ wt,
                        int K, int Nc) {
  __shared__ float t[64][65];          // +1 pad: no bank conflicts on transpose
  const int k0 = blockIdx.x * 64;
  const int n0 = blockIdx.y * 64;
  const int tid = threadIdx.x;
#pragma unroll
  for (int i = 0; i < 16; ++i) {
    int e = tid + 256 * i;
    int k = e >> 6, n = e & 63;
    t[k][n] = w[(size_t)(k0 + k) * Nc + n0 + n];
  }
  __syncthreads();
#pragma unroll
  for (int i = 0; i < 16; ++i) {
    int e = tid + 256 * i;
    int n = e >> 6, k = e & 63;
    wt[(size_t)(n0 + n) * K + k0 + k] = (__bf16)t[k][n];
  }
}

// ---------------------------------------------------------------------------
// LayerNorm over C=768: one 256-thread block per row. wave32 shuffle
// reductions + single 8-partial LDS combine (one barrier).
// ---------------------------------------------------------------------------
__global__ __launch_bounds__(256)
void ln_kernel(const float* __restrict__ x, const float* __restrict__ g,
               const float* __restrict__ b, __bf16* __restrict__ y) {
  __shared__ float ps[8], ps2[8];
  const int row  = blockIdx.x;
  const int tid  = threadIdx.x;
  const int wave = tid >> 5;
  const int lane = tid & 31;
  const float* xr = x + (size_t)row * Cc;

  float v[3];
  float s = 0.f, s2 = 0.f;
#pragma unroll
  for (int i = 0; i < 3; ++i) {
    v[i] = xr[tid + 256 * i];
    s  += v[i];
    s2 += v[i] * v[i];
  }
#pragma unroll
  for (int msk = 16; msk >= 1; msk >>= 1) {
    s  += __shfl_xor(s,  msk, 32);
    s2 += __shfl_xor(s2, msk, 32);
  }
  if (lane == 0) { ps[wave] = s; ps2[wave] = s2; }
  __syncthreads();
  float ts = 0.f, ts2 = 0.f;
#pragma unroll
  for (int w = 0; w < 8; ++w) { ts += ps[w]; ts2 += ps2[w]; }
  const float mean = ts * (1.0f / Cc);
  const float var  = ts2 * (1.0f / Cc) - mean * mean;
  const float rstd = rsqrtf(var + 1e-5f);

#pragma unroll
  for (int i = 0; i < 3; ++i) {
    int c = tid + 256 * i;
    y[(size_t)row * Cc + c] = (__bf16)((v[i] - mean) * rstd * g[c] + b[c]);
  }
}

// ---------------------------------------------------------------------------
// WMMA GEMM: out[M,Nc] = A[M,K](bf16) @ Wt[Nc,K](bf16, pre-transposed)
// + compile-time fused epilogue.
// Block tile 128x128, BK=64. 8 waves in 2(M)x4(N); each wave owns 64x32
// (4x2 WMMA tiles). Both A and Wt tiles staged with async b128 copies.
// ---------------------------------------------------------------------------
template <int HAS_BIAS, int DO_GELU, int HAS_RES, int OUT_BF, int OUT_F>
__global__ __launch_bounds__(256)
void gemm_bf16_wmma(const __bf16* __restrict__ A, const __bf16* __restrict__ Wt,
                    int M, int K, int Nc,
                    const float* __restrict__ bias,
                    const float* __restrict__ res, const float* __restrict__ ls,
                    __bf16* __restrict__ out_bf, float* __restrict__ out_f) {
  __shared__ __attribute__((aligned(16))) __bf16 sA[128 * 64];   // [m][k]
  __shared__ __attribute__((aligned(16))) __bf16 sW[128 * 64];   // [n][k]

  const int m0 = blockIdx.x * 128;
  const int n0 = blockIdx.y * 128;
  const int tid  = threadIdx.x;
  const int wave = tid >> 5;
  const int lane = tid & 31;
  const int half = lane >> 4;
  const int l16  = lane & 15;
  const int wm = wave >> 2;     // 0..1 -> 64-row band
  const int wn = wave & 3;      // 0..3 -> 32-col band

  v8f acc[4][2] = {};

  for (int k0 = 0; k0 < K; k0 += 64) {
    // Stage A tile 128x64 bf16 = 1024 b128 chunks (4 per thread).
#pragma unroll
    for (int i = 0; i < 4; ++i) {
      int e = tid + 256 * i;
      int r = e >> 3, c = (e & 7) * 8;
      async_b128(sA + r * 64 + c, A + (size_t)(m0 + r) * K + k0 + c);
    }
    // Stage Wt tile 128x64 bf16 = 1024 b128 chunks (4 per thread).
#pragma unroll
    for (int i = 0; i < 4; ++i) {
      int e = tid + 256 * i;
      int r = e >> 3, c = (e & 7) * 8;
      async_b128(sW + r * 64 + c, Wt + (size_t)(n0 + r) * K + k0 + c);
    }
    async_wait0();
    __syncthreads();

#pragma unroll
    for (int ks = 0; ks < 64; ks += 32) {
      v16bf af[4], bfv[2];
#pragma unroll
      for (int im = 0; im < 4; ++im)
        af[im] = frag_ld(sA, wm * 64 + im * 16 + l16, 64, ks, half);
#pragma unroll
      for (int in = 0; in < 2; ++in)
        bfv[in] = frag_ld(sW, wn * 32 + in * 16 + l16, 64, ks, half);
#pragma unroll
      for (int im = 0; im < 4; ++im)
#pragma unroll
        for (int in = 0; in < 2; ++in)
          acc[im][in] = wmma_bf16(af[im], bfv[in], acc[im][in]);
    }
    __syncthreads();
  }

  // Compile-time fused epilogue: bias -> (GELU) -> (residual+LayerScale).
#pragma unroll
  for (int im = 0; im < 4; ++im)
#pragma unroll
    for (int in = 0; in < 2; ++in)
#pragma unroll
      for (int r = 0; r < 8; ++r) {
        int row = m0 + wm * 64 + im * 16 + r + 8 * half;
        int col = n0 + wn * 32 + in * 16 + l16;
        float v = acc[im][in][r];
        if (HAS_BIAS) v += bias[col];
        if (DO_GELU)  v = 0.5f * v * (1.0f + erff(v * 0.70710678118654752f));
        size_t idx = (size_t)row * Nc + col;
        if (HAS_RES)  v = res[idx] + ls[col] * v;
        if (OUT_F)    out_f[idx]  = v;
        if (OUT_BF)   out_bf[idx] = (__bf16)v;
      }
}

// ---------------------------------------------------------------------------
// Flash attention: one block per (row-tile of 128, head, batch).
// 8 waves, each owns 16 query rows. Online softmax kept in registers using
// half-wave xor shuffles. Q/K tiles staged with async b128 copies.
// ---------------------------------------------------------------------------
__global__ __launch_bounds__(256)
void attn_kernel(const __bf16* __restrict__ qkv, __bf16* __restrict__ ob) {
  __shared__ __attribute__((aligned(16))) __bf16 sQ[128 * 64];  // [m][d]
  __shared__ __attribute__((aligned(16))) __bf16 sK[64 * 64];   // [j][d]
  __shared__ __attribute__((aligned(16))) __bf16 sVt[64 * 64];  // [d][j]
  __shared__ __attribute__((aligned(16))) __bf16 sP[128 * 64];  // [m][j] bf16

  const int rt = blockIdx.x;   // row tile (0..7), 128 rows each
  const int h  = blockIdx.y;   // head
  const int bb = blockIdx.z;   // batch
  const int tid  = threadIdx.x;
  const int wave = tid >> 5;
  const int lane = tid & 31;
  const int half = lane >> 4;
  const int l16  = lane & 15;

  const int rowE = 3 * Cc;          // 2304 bf16 per token row

  // Stage Q tile (128 rows x 64 bf16 = 1024 b128 chunks; 4 per thread).
#pragma unroll
  for (int i = 0; i < 4; ++i) {
    int e = tid + 256 * i;
    int r = e >> 3, c = (e & 7) * 8;
    async_b128(sQ + r * 64 + c,
               qkv + (size_t)(bb * Nn + rt * 128 + r) * rowE + h * HD + c);
  }
  async_wait0();
  __syncthreads();

  v16bf aQ[2];
  aQ[0] = frag_ld(sQ, wave * 16 + l16, 64, 0,  half);
  aQ[1] = frag_ld(sQ, wave * 16 + l16, 64, 32, half);

  v8f  o_acc[4] = {};
  float mrow[8], lrow[8];
#pragma unroll
  for (int r = 0; r < 8; ++r) { mrow[r] = -3.0e38f; lrow[r] = 0.f; }

  for (int jt = 0; jt < Nn / 64; ++jt) {
    // Stage K tile [j][d] (512 b128 chunks; 2 per thread, async).
#pragma unroll
    for (int i = 0; i < 2; ++i) {
      int e = tid + 256 * i;
      int r = e >> 3, c = (e & 7) * 8;
      async_b128(sK + r * 64 + c,
                 qkv + (size_t)(bb * Nn + jt * 64 + r) * rowE + Cc + h * HD + c);
    }
    // Stage V transposed [d][j] (transpose -> scalar path, 16 per thread).
#pragma unroll
    for (int i = 0; i < 16; ++i) {
      int e = tid + 256 * i;
      int j = e >> 6, d = e & 63;
      sVt[d * 64 + j] =
          qkv[(size_t)(bb * Nn + jt * 64 + j) * rowE + 2 * Cc + h * HD + d];
    }
    async_wait0();
    __syncthreads();

    // S = Q K^T (scale applied below) : 4 col tiles, K-dim 64 in two WMMA steps.
    v8f s[4] = {};
#pragma unroll
    for (int kk = 0; kk < 2; ++kk)
#pragma unroll
      for (int nt = 0; nt < 4; ++nt) {
        v16bf bK = frag_ld(sK, nt * 16 + l16, 64, kk * 32, half);
        s[nt] = wmma_bf16(aQ[kk], bK, s[nt]);
      }

    // Online softmax per owned row (row = wave*16 + r + 8*half).
#pragma unroll
    for (int r = 0; r < 8; ++r) {
      float mx = -3.0e38f;
#pragma unroll
      for (int nt = 0; nt < 4; ++nt) {
        s[nt][r] *= 0.125f;                     // 1/sqrt(HD)
        mx = fmaxf(mx, s[nt][r]);
      }
#pragma unroll
      for (int msk = 8; msk >= 1; msk >>= 1)
        mx = fmaxf(mx, __shfl_xor(mx, msk, 32));
      float mnew  = fmaxf(mrow[r], mx);
      float alpha = __expf(mrow[r] - mnew);
      float rs = 0.f;
#pragma unroll
      for (int nt = 0; nt < 4; ++nt) {
        float pv = __expf(s[nt][r] - mnew);
        s[nt][r] = pv; rs += pv;
      }
#pragma unroll
      for (int msk = 8; msk >= 1; msk >>= 1)
        rs += __shfl_xor(rs, msk, 32);
      lrow[r] = lrow[r] * alpha + rs;
      mrow[r] = mnew;
#pragma unroll
      for (int nt = 0; nt < 4; ++nt) o_acc[nt][r] *= alpha;
      // Spill P row as bf16 to this wave's private LDS band.
#pragma unroll
      for (int nt = 0; nt < 4; ++nt)
        sP[(wave * 16 + r + 8 * half) * 64 + nt * 16 + l16] = (__bf16)s[nt][r];
    }

    // O += P @ V  (same-wave LDS RAW: DS ops are in-order per wave).
#pragma unroll
    for (int kk = 0; kk < 2; ++kk) {
      v16bf aP = frag_ld(sP, wave * 16 + l16, 64, kk * 32, half);
#pragma unroll
      for (int nt = 0; nt < 4; ++nt) {
        v16bf bV = frag_ld(sVt, nt * 16 + l16, 64, kk * 32, half);
        o_acc[nt] = wmma_bf16(aP, bV, o_acc[nt]);
      }
    }
    __syncthreads();
  }

  // Normalize and store O as bf16 into [B,N,C] (c = h*HD + d).
#pragma unroll
  for (int r = 0; r < 8; ++r) {
    float inv = 1.0f / lrow[r];
    int nrow = rt * 128 + wave * 16 + r + 8 * half;
    size_t base = ((size_t)(bb * Nn + nrow)) * Cc + h * HD;
#pragma unroll
    for (int nt = 0; nt < 4; ++nt)
      ob[base + nt * 16 + l16] = (__bf16)(o_acc[nt][r] * inv);
  }
}

// ---------------------------------------------------------------------------
// Host-side orchestration.
// ---------------------------------------------------------------------------
extern "C" void kernel_launch(void* const* d_in, const int* in_sizes, int n_in,
                              void* d_out, int out_size, void* d_ws, size_t ws_size,
                              hipStream_t stream) {
  const float* x      = (const float*)d_in[0];
  const float* w_qkv  = (const float*)d_in[1];
  const float* w_proj = (const float*)d_in[2];
  const float* b_proj = (const float*)d_in[3];
  const float* ln1_g  = (const float*)d_in[4];
  const float* ln1_b  = (const float*)d_in[5];
  const float* ln2_g  = (const float*)d_in[6];
  const float* ln2_b  = (const float*)d_in[7];
  const float* ls1_g  = (const float*)d_in[8];
  const float* ls2_g  = (const float*)d_in[9];
  const float* w_fc1  = (const float*)d_in[10];
  const float* b_fc1  = (const float*)d_in[11];
  const float* w_fc2  = (const float*)d_in[12];
  const float* b_fc2  = (const float*)d_in[13];
  float* out = (float*)d_out;

  char* ws = (char*)d_ws;
  size_t off = 0;
  auto carve = [&](size_t bytes) -> void* {
    void* p = ws + off;
    off = (off + bytes + 255) & ~(size_t)255;
    return p;
  };
  __bf16* yb    = (__bf16*)carve((size_t)Mrows * Cc * 2);        // LN out (reused)
  __bf16* qkvb  = (__bf16*)carve((size_t)Mrows * 3 * Cc * 2);    // QKV bf16
  __bf16* obuf  = (__bf16*)carve((size_t)Mrows * Cc * 2);        // attn out bf16
  float*  x1    = (float*) carve((size_t)Mrows * Cc * 4);        // residual 1
  __bf16* h1b   = (__bf16*)carve((size_t)Mrows * MLPd * 2);      // GELU(fc1) bf16
  __bf16* wqb   = (__bf16*)carve((size_t)Cc * 3 * Cc * 2);       // [3C][C]
  __bf16* wpb   = (__bf16*)carve((size_t)Cc * Cc * 2);           // [C][C]
  __bf16* w1b   = (__bf16*)carve((size_t)Cc * MLPd * 2);         // [MLP][C]
  __bf16* w2b   = (__bf16*)carve((size_t)MLPd * Cc * 2);         // [C][MLP]

  // Weight convert + transpose fp32[K][N] -> bf16[N][K].
  cvt_transpose_bf16<<<dim3(Cc / 64, (3 * Cc) / 64), 256, 0, stream>>>(
      w_qkv, wqb, Cc, 3 * Cc);
  cvt_transpose_bf16<<<dim3(Cc / 64, Cc / 64), 256, 0, stream>>>(
      w_proj, wpb, Cc, Cc);
  cvt_transpose_bf16<<<dim3(Cc / 64, MLPd / 64), 256, 0, stream>>>(
      w_fc1, w1b, Cc, MLPd);
  cvt_transpose_bf16<<<dim3(MLPd / 64, Cc / 64), 256, 0, stream>>>(
      w_fc2, w2b, MLPd, Cc);

  // --- attention branch ---
  ln_kernel<<<Mrows, 256, 0, stream>>>(x, ln1_g, ln1_b, yb);

  // qkv = LN1(x) @ w_qkv   (no bias in reference)
  gemm_bf16_wmma<0, 0, 0, 1, 0><<<dim3(Mrows / 128, (3 * Cc) / 128), 256, 0, stream>>>(
      yb, wqb, Mrows, Cc, 3 * Cc,
      nullptr, nullptr, nullptr, qkvb, nullptr);

  attn_kernel<<<dim3(Nn / 128, Hh, Bq), 256, 0, stream>>>(qkvb, obuf);

  // x1 = x + ls1 * (o @ w_proj + b_proj)
  gemm_bf16_wmma<1, 0, 1, 0, 1><<<dim3(Mrows / 128, Cc / 128), 256, 0, stream>>>(
      obuf, wpb, Mrows, Cc, Cc,
      b_proj, x, ls1_g, nullptr, x1);

  // --- MLP branch ---
  ln_kernel<<<Mrows, 256, 0, stream>>>(x1, ln2_g, ln2_b, yb);

  // h1 = gelu(y2 @ w_fc1 + b_fc1)
  gemm_bf16_wmma<1, 1, 0, 1, 0><<<dim3(Mrows / 128, MLPd / 128), 256, 0, stream>>>(
      yb, w1b, Mrows, Cc, MLPd,
      b_fc1, nullptr, nullptr, h1b, nullptr);

  // out = x1 + ls2 * (h1 @ w_fc2 + b_fc2)
  gemm_bf16_wmma<1, 0, 1, 0, 1><<<dim3(Mrows / 128, Cc / 128), 256, 0, stream>>>(
      h1b, w2b, Mrows, MLPd, Cc,
      b_fc2, x1, ls2_g, nullptr, out);
}
